// GCNN_mutual_attention_25872882991628
// MI455X (gfx1250) — matrix operand order repaired and checked by reading, not compiled, not run on hardware
//
#include <hip/hip_runtime.h>
#include <hip/hip_bf16.h>
#include <cstdint>
#include <cstddef>

// ---------------- problem constants (match reference) ----------------
#define NNODES 16000
#define Bc     32
#define NPGc   500
#define Ec     256000
#define Fc     1024
#define ODc    128
#define DDc    80
#define TDc    32
#define NHc    4
#define DHc    8
#define DFFc   128
#define NLc    2
#define LSUBc  128
#define MAXLENc 512
#define NEGc   (-1e9f)
#define SLOPEc 0.01f
#define EPSc   1e-5f

// ---------------- bf16 helpers ----------------
typedef __attribute__((ext_vector_type(16))) __bf16 v16bf;
typedef __attribute__((ext_vector_type(8)))  float  v8f;

__device__ __forceinline__ __bf16 us2bf(unsigned short u) {
  union { unsigned short s; __bf16 b; } c; c.s = u; return c.b;
}
__device__ __forceinline__ unsigned short f2us(float f) {
  unsigned u = __float_as_uint(f);
  u += 0x7FFFu + ((u >> 16) & 1u);     // round-to-nearest-even
  return (unsigned short)(u >> 16);
}
__device__ __forceinline__ float us2f(unsigned short u) {
  return __uint_as_float(((unsigned)u) << 16);
}

// ---------------- generic convert f32 -> bf16 ----------------
__global__ void k_cvt_bf16(const float* __restrict__ in, unsigned short* __restrict__ out, size_t n) {
  size_t i = (size_t)blockIdx.x * blockDim.x + threadIdx.x;
  if (i < n) out[i] = f2us(in[i]);
}

// ---------------- WMMA GEMM: H(bf16) = X(bf16)[M,K] @ W(bf16)[Ncols,K]^T ----------------
// One wave computes a 16x64 output tile (4 wmma accumulators), K-loop step 32.
// Fragments use contiguous k = 16*half + e for both A and B (wave32, 16-bit A 16x32; B 32x16).
__global__ __launch_bounds__(128) void k_gemm_bf16(
    const unsigned short* __restrict__ X,   // [M,K]
    const unsigned short* __restrict__ W,   // [Ncols,K] row-major (B[k][j] = W[j*K+k])
    unsigned short* __restrict__ H,         // [M,Ncols]
    int M, int K, int Ncols)
{
  const int wave = threadIdx.x >> 5;
  const int lane = threadIdx.x & 31;
  const int half = lane >> 4;
  const int col  = lane & 15;

  const int m0 = blockIdx.x * 16;
  const int n0 = (blockIdx.y * 4 + wave) * 64;          // 64 cols per wave
  const int m  = m0 + col;                              // A: M = lane%16

  v8f acc[4];
  #pragma unroll
  for (int t = 0; t < 4; ++t) acc[t] = (v8f){0.f,0.f,0.f,0.f,0.f,0.f,0.f,0.f};

  for (int kk = 0; kk < K; kk += 32) {
    // ---- A fragment: 16x32 bf16 ----
    v16bf a;
    {
      const unsigned short* ap0 = X + (size_t)m * K + kk + half * 16;       // K = kk + half*16 + e
      const unsigned short* ap1 = X + (size_t)m * K + kk + half * 16 + 8;   // K = kk+half*16+8+e
      #pragma unroll
      for (int e = 0; e < 8; ++e) { a[e] = us2bf(ap0[e]); a[8 + e] = us2bf(ap1[e]); }
      if (kk + 32 < K) __builtin_prefetch(X + (size_t)m * K + kk + 32, 0, 3);
    }
    // ---- 4 B fragments + 4 WMMAs, reusing A ----
    #pragma unroll
    for (int t = 0; t < 4; ++t) {
      const int j = n0 + t * 16 + col;                  // B column = lane%16
      const unsigned short* bp = W + (size_t)j * K + kk + half * 16;        // K = kk + half*16 + e
      v16bf b;
      #pragma unroll
      for (int e = 0; e < 16; ++e) b[e] = us2bf(bp[e]);
      acc[t] = __builtin_amdgcn_wmma_f32_16x16x32_bf16(
          /*neg_a=*/false, a, /*neg_b=*/false, b,
          /*c_mod=*/(short)0, acc[t], /*reuse_a=*/false, /*reuse_b=*/false);
    }
  }

  // ---- store D (f32 16x16 layout: VGPR v -> row v + 8*half; col = lane%16), as bf16 ----
  #pragma unroll
  for (int t = 0; t < 4; ++t) {
    const int j = n0 + t * 16 + col;
    #pragma unroll
    for (int v = 0; v < 8; ++v) {
      const int row = m0 + v + 8 * half;
      H[(size_t)row * Ncols + j] = f2us(acc[t][v]);
    }
  }
}

// ---------------- degree: deg[n] = 1 (self loop) + in-degree ----------------
__global__ void k_deg_init(float* __restrict__ deg, int n) {
  int i = blockIdx.x * blockDim.x + threadIdx.x;
  if (i < n) deg[i] = 1.0f;
}
__global__ void k_deg_accum(const int* __restrict__ ei, float* __restrict__ deg, int ne) {
  int e = blockIdx.x * blockDim.x + threadIdx.x;
  if (e < ne) atomicAdd(&deg[ei[ne + e]], 1.0f);        // dst row of edge_index
}

// ---------------- agg init with self-loop: agg = h * (1/deg) ----------------
__global__ void k_agg_selfloop(const unsigned short* __restrict__ h, const float* __restrict__ deg,
                               float* __restrict__ agg, size_t n, int F) {
  size_t i = (size_t)blockIdx.x * blockDim.x + threadIdx.x;
  if (i < n) {
    int node = (int)(i / F);
    agg[i] = us2f(h[i]) / deg[node];                    // dinv[n]*dinv[n] = 1/deg
  }
}

// ---------------- edge scatter: agg[dst] += h[src] * dinv[src]*dinv[dst] ----------------
__global__ __launch_bounds__(256) void k_edge_scatter(
    const int* __restrict__ ei, const unsigned short* __restrict__ h,
    const float* __restrict__ deg, float* __restrict__ agg, int ne, int F)
{
  __shared__ float s_norm;
  __shared__ int   s_src, s_dst;
  const int e = blockIdx.x;
  if (threadIdx.x == 0) {
    int src = ei[e], dst = ei[ne + e];
    s_src = src; s_dst = dst;
    s_norm = rsqrtf(deg[src]) * rsqrtf(deg[dst]);
  }
  __syncthreads();
  const unsigned short* hp = h + (size_t)s_src * F;
  float* ap = agg + (size_t)s_dst * F;
  const float norm = s_norm;
  for (int f = threadIdx.x; f < F; f += 256)
    atomicAdd(&ap[f], us2f(hp[f]) * norm);
}

// ---------------- leaky(agg + b) then mean-pool per contiguous graph ----------------
__global__ void k_pool(const float* __restrict__ agg, const float* __restrict__ bias,
                       float* __restrict__ pooled, int F) {
  int g = blockIdx.y;
  int f = blockIdx.x * blockDim.x + threadIdx.x;
  float b = bias[f];
  float s = 0.f;
  size_t base = (size_t)g * NPGc;
  for (int i = 0; i < NPGc; ++i) {
    float v = agg[(base + i) * F + f] + b;
    s += (v >= 0.f) ? v : SLOPEc * v;
  }
  pooled[(size_t)g * F + f] = s * (1.0f / NPGc);
}

// ---------------- x_out = leaky(pooled @ fcW^T + fcB) ----------------
__global__ void k_fc_leaky(const float* __restrict__ pooled, const float* __restrict__ w,
                           const float* __restrict__ b, float* __restrict__ out, int F, int OD) {
  int i = blockIdx.x * blockDim.x + threadIdx.x;
  if (i >= Bc * OD) return;
  int bb = i / OD, o = i % OD;
  const float* wp = w + (size_t)o * F;
  const float* pp = pooled + (size_t)bb * F;
  float s = b[o];
  for (int k = 0; k < F; ++k) s += pp[k] * wp[k];
  out[i] = (s >= 0.f) ? s : SLOPEc * s;
}

// ---------------- zero a float buffer ----------------
__global__ void k_zero(float* __restrict__ p, size_t n) {
  size_t i = (size_t)blockIdx.x * blockDim.x + threadIdx.x;
  if (i < n) p[i] = 0.f;
}

// ---------------- total lengths per batch ----------------
__global__ void k_total_len(const int* l0, const int* l1, const int* l2, const int* l3, int* tot) {
  int b = threadIdx.x;
  if (b < Bc) tot[b] = l0[b] + l1[b] + l2[b] + l3[b];
}

// ---------------- ragged pack: project (DD->30), append indicators, scatter at offsets ----------------
__global__ __launch_bounds__(LSUBc) void k_seq_pack(
    const float* m0, const float* m1, const float* m2, const float* m3,
    const int* l0, const int* l1, const int* l2, const int* l3,
    const float* __restrict__ red_w, const float* __restrict__ red_b,
    float* __restrict__ seq)
{
  const int b   = blockIdx.x & (Bc - 1);
  const int seg = blockIdx.x >> 5;
  const int p   = threadIdx.x;

  const float* Ms[4] = { m0, m1, m2, m3 };
  const int*   Ls[4] = { l0, l1, l2, l3 };
  const float  SI[4] = { 1.f, 0.f, 1.f, 0.f };
  const float  FI[4] = { 1.f, 1.f, 0.f, 0.f };

  int off = 0;
  for (int s = 0; s < seg; ++s) off += Ls[s][b];
  const int len = Ls[seg][b];
  if (p >= len) return;

  const float* src = Ms[seg] + ((size_t)b * LSUBc + p) * DDc;
  float* dst = seq + ((size_t)b * MAXLENc + off + p) * TDc;
  for (int j = 0; j < TDc - 2; ++j) {
    const float* wp = red_w + (size_t)j * DDc;
    float s = red_b[j];
    for (int k = 0; k < DDc; ++k) s += src[k] * wp[k];
    dst[j] = s;
  }
  dst[TDc - 2] = SI[seg];
  dst[TDc - 1] = FI[seg];
}

// ---------------- qkv = x @ Win^T + bin  (Win [96,32]) ----------------
__global__ void k_qkv(const float* __restrict__ xt, const float* __restrict__ w,
                      const float* __restrict__ b, float* __restrict__ qkv) {
  int i = blockIdx.x * blockDim.x + threadIdx.x;
  if (i >= Bc * MAXLENc * 3 * TDc) return;
  int j = i % (3 * TDc);
  int row = i / (3 * TDc);
  const float* xp = xt + (size_t)row * TDc;
  const float* wp = w + (size_t)j * TDc;
  float s = b[j];
  #pragma unroll
  for (int k = 0; k < TDc; ++k) s += xp[k] * wp[k];
  qkv[i] = s;
}

// ---------------- attention: one wave per (b, head, query); key-padding masked softmax ----------------
__global__ __launch_bounds__(128) void k_attn(const float* __restrict__ qkv,
                                              const int* __restrict__ tot,
                                              float* __restrict__ o) {
  const int wid  = blockIdx.x * (blockDim.x >> 5) + (threadIdx.x >> 5);
  const int lane = threadIdx.x & 31;
  const int q   = wid % MAXLENc;
  int t         = wid / MAXLENc;
  const int h   = t % NHc;
  const int b   = t / NHc;

  const int L = tot[b];
  const float* base = qkv + (size_t)b * MAXLENc * (3 * TDc);

  float qv[DHc];
  const float* qp = base + (size_t)q * (3 * TDc) + h * DHc;
  #pragma unroll
  for (int d = 0; d < DHc; ++d) qv[d] = qp[d];

  const float scale = 0.35355339059327373f;     // 1/sqrt(8)
  float sc[MAXLENc / 32];
  float smax = -3.0e38f;
  #pragma unroll
  for (int c = 0; c < MAXLENc / 32; ++c) {
    const int k = c * 32 + lane;
    const float* kp = base + (size_t)k * (3 * TDc) + TDc + h * DHc;
    float s = 0.f;
    #pragma unroll
    for (int d = 0; d < DHc; ++d) s += qv[d] * kp[d];
    s *= scale;
    if (k >= L) s = NEGc;
    sc[c] = s;
    smax = fmaxf(smax, s);
  }
  #pragma unroll
  for (int off = 16; off > 0; off >>= 1) smax = fmaxf(smax, __shfl_xor(smax, off, 32));

  float ssum = 0.f;
  float acc[DHc];
  #pragma unroll
  for (int d = 0; d < DHc; ++d) acc[d] = 0.f;
  #pragma unroll
  for (int c = 0; c < MAXLENc / 32; ++c) {
    const int k = c * 32 + lane;
    const float p = expf(sc[c] - smax);
    ssum += p;
    const float* vp = base + (size_t)k * (3 * TDc) + 2 * TDc + h * DHc;
    #pragma unroll
    for (int d = 0; d < DHc; ++d) acc[d] += p * vp[d];
  }
  #pragma unroll
  for (int off = 16; off > 0; off >>= 1) {
    ssum += __shfl_xor(ssum, off, 32);
    #pragma unroll
    for (int d = 0; d < DHc; ++d) acc[d] += __shfl_xor(acc[d], off, 32);
  }
  if (lane == 0) {
    const float inv = 1.0f / ssum;
    float* op = o + ((size_t)b * MAXLENc + q) * TDc + h * DHc;
    #pragma unroll
    for (int d = 0; d < DHc; ++d) op[d] = acc[d] * inv;
  }
}

// ---------------- out-proj + residual + LayerNorm (one thread per row) ----------------
__global__ void k_attnout_ln(const float* __restrict__ xt_in, const float* __restrict__ o,
                             const float* __restrict__ wout, const float* __restrict__ bout,
                             const float* __restrict__ lnw, const float* __restrict__ lnb,
                             float* __restrict__ xt_out) {
  int row = blockIdx.x * blockDim.x + threadIdx.x;
  if (row >= Bc * MAXLENc) return;
  const float* op = o + (size_t)row * TDc;
  const float* xp = xt_in + (size_t)row * TDc;
  float r[TDc];
  #pragma unroll
  for (int t = 0; t < TDc; ++t) {
    const float* wp = wout + (size_t)t * TDc;
    float s = bout[t];
    #pragma unroll
    for (int k = 0; k < TDc; ++k) s += op[k] * wp[k];
    r[t] = xp[t] + s;
  }
  float m = 0.f;
  #pragma unroll
  for (int t = 0; t < TDc; ++t) m += r[t];
  m *= (1.0f / TDc);
  float v = 0.f;
  #pragma unroll
  for (int t = 0; t < TDc; ++t) { float d = r[t] - m; v += d * d; }
  v *= (1.0f / TDc);
  const float inv = rsqrtf(v + EPSc);
  float* out = xt_out + (size_t)row * TDc;
  #pragma unroll
  for (int t = 0; t < TDc; ++t) out[t] = (r[t] - m) * inv * lnw[t] + lnb[t];
}

// ---------------- FFN (relu) + residual + LayerNorm (one 128-thread block per row) ----------------
__global__ __launch_bounds__(DFFc) void k_ffn_ln(
    const float* __restrict__ xt_in,
    const float* __restrict__ w1, const float* __restrict__ b1,   // [128,32],[128]
    const float* __restrict__ w2, const float* __restrict__ b2,   // [32,128],[32]
    const float* __restrict__ lnw, const float* __restrict__ lnb,
    float* __restrict__ xt_out)
{
  __shared__ float hid[DFFc];
  __shared__ float xrow[TDc];
  __shared__ float rrow[TDc];
  __shared__ float red[2];
  const int row = blockIdx.x;
  const int tid = threadIdx.x;
  const float* xp = xt_in + (size_t)row * TDc;
  if (tid < TDc) xrow[tid] = xp[tid];
  __syncthreads();
  {
    const float* wp = w1 + (size_t)tid * TDc;
    float s = b1[tid];
    #pragma unroll
    for (int k = 0; k < TDc; ++k) s += xrow[k] * wp[k];
    hid[tid] = (s > 0.f) ? s : 0.f;
  }
  __syncthreads();
  if (tid < TDc) {
    const float* wp = w2 + (size_t)tid * DFFc;
    float s = b2[tid];
    #pragma unroll
    for (int j = 0; j < DFFc; ++j) s += hid[j] * wp[j];
    rrow[tid] = xrow[tid] + s;
  }
  __syncthreads();
  if (tid == 0) {
    float m = 0.f;
    for (int t = 0; t < TDc; ++t) m += rrow[t];
    m *= (1.0f / TDc);
    float v = 0.f;
    for (int t = 0; t < TDc; ++t) { float d = rrow[t] - m; v += d * d; }
    v *= (1.0f / TDc);
    red[0] = m; red[1] = rsqrtf(v + EPSc);
  }
  __syncthreads();
  if (tid < TDc)
    xt_out[(size_t)row * TDc + tid] = (rrow[tid] - red[0]) * red[1] * lnw[tid] + lnb[tid];
}

// ---------------- masked mean over valid positions ----------------
__global__ void k_masked_mean(const float* __restrict__ xt, const int* __restrict__ tot,
                              float* __restrict__ mas) {
  int i = blockIdx.x * blockDim.x + threadIdx.x;
  if (i >= Bc * TDc) return;
  int b = i / TDc, t = i % TDc;
  int L = tot[b];
  float s = 0.f;
  for (int p = 0; p < L; ++p) s += xt[((size_t)b * MAXLENc + p) * TDc + t];
  mas[i] = s / (float)L;
}

// ---------------- final: concat [x1,x2,mas] @ final_w^T + final_b ----------------
__global__ void k_final(const float* __restrict__ x1, const float* __restrict__ x2,
                        const float* __restrict__ mas, const float* __restrict__ fw,
                        const float* __restrict__ fb, float* __restrict__ out) {
  int b = blockIdx.x * blockDim.x + threadIdx.x;
  if (b >= Bc) return;
  float s = fb[0];
  for (int k = 0; k < ODc; ++k) s += x1[b * ODc + k] * fw[k];
  for (int k = 0; k < ODc; ++k) s += x2[b * ODc + k] * fw[ODc + k];
  for (int k = 0; k < TDc;  ++k) s += mas[b * TDc + k] * fw[2 * ODc + k];
  out[b] = s;
}

// =====================================================================
extern "C" void kernel_launch(void* const* d_in, const int* in_sizes, int n_in,
                              void* d_out, int out_size, void* d_ws, size_t ws_size,
                              hipStream_t stream) {
  (void)in_sizes; (void)n_in; (void)out_size; (void)ws_size;

  // ---- inputs (setup_inputs dict order) ----
  const float* pro1_x = (const float*)d_in[0];
  const float* pro2_x = (const float*)d_in[1];
  const int*   p1_ei  = (const int*)d_in[2];
  const int*   p2_ei  = (const int*)d_in[3];
  // d_in[4], d_in[5]: batch vectors (structure is contiguous repeat -> unused)
  const float* m1s = (const float*)d_in[6];   const int* l1s = (const int*)d_in[7];
  const float* m1f = (const float*)d_in[8];   const int* l1f = (const int*)d_in[9];
  const float* m2s = (const float*)d_in[10];  const int* l2s = (const int*)d_in[11];
  const float* m2f = (const float*)d_in[12];  const int* l2f = (const int*)d_in[13];
  const float* gcn1_w = (const float*)d_in[14]; const float* gcn1_b = (const float*)d_in[15];
  const float* gcn2_w = (const float*)d_in[16]; const float* gcn2_b = (const float*)d_in[17];
  const float* fc1_w  = (const float*)d_in[18]; const float* fc1_b  = (const float*)d_in[19];
  const float* fc2_w  = (const float*)d_in[20]; const float* fc2_b  = (const float*)d_in[21];
  const float* red_w  = (const float*)d_in[22]; const float* red_b  = (const float*)d_in[23];
  const float* ain_w  = (const float*)d_in[24]; const float* ain_b  = (const float*)d_in[25];
  const float* aout_w = (const float*)d_in[26]; const float* aout_b = (const float*)d_in[27];
  const float* ln1_w  = (const float*)d_in[28]; const float* ln1_b  = (const float*)d_in[29];
  const float* ln2_w  = (const float*)d_in[30]; const float* ln2_b  = (const float*)d_in[31];
  const float* ff1_w  = (const float*)d_in[32]; const float* ff1_b  = (const float*)d_in[33];
  const float* ff2_w  = (const float*)d_in[34]; const float* ff2_b  = (const float*)d_in[35];
  const float* fin_w  = (const float*)d_in[36]; const float* fin_b  = (const float*)d_in[37];
  float* out = (float*)d_out;

  // ---- carve workspace (byte offsets, 256-aligned) ----
  char* ws = (char*)d_ws;
  size_t cur = 0;
  auto carve = [&](size_t bytes) -> char* {
    char* p = ws + cur;
    cur += (bytes + 255) & ~(size_t)255;
    return p;
  };
  unsigned short* xb16  = (unsigned short*)carve((size_t)NNODES * Fc * 2);
  unsigned short* wb16  = (unsigned short*)carve((size_t)Fc * Fc * 2);
  unsigned short* h16   = (unsigned short*)carve((size_t)NNODES * Fc * 2);
  float* agg    = (float*)carve((size_t)NNODES * Fc * 4);
  float* deg    = (float*)carve((size_t)NNODES * 4);
  float* pooled = (float*)carve((size_t)Bc * Fc * 4);
  float* x1     = (float*)carve((size_t)Bc * ODc * 4);
  float* x2     = (float*)carve((size_t)Bc * ODc * 4);
  float* xtA    = (float*)carve((size_t)Bc * MAXLENc * TDc * 4);
  float* xtB    = (float*)carve((size_t)Bc * MAXLENc * TDc * 4);
  float* qkv    = (float*)carve((size_t)Bc * MAXLENc * 3 * TDc * 4);
  float* attno  = (float*)carve((size_t)Bc * MAXLENc * TDc * 4);
  int*   tot    = (int*)carve((size_t)Bc * 4);
  float* masout = (float*)carve((size_t)Bc * TDc * 4);

  const size_t nx = (size_t)NNODES * Fc;

  // ================= GCN branches (scratch reused, stream-ordered) =================
  auto run_branch = [&](const float* x, const int* ei, const float* gw, const float* gb,
                        const float* fw, const float* fb, float* xout) {
    k_cvt_bf16<<<(unsigned)((nx + 255) / 256), 256, 0, stream>>>(x, xb16, nx);
    k_cvt_bf16<<<(unsigned)(((size_t)Fc * Fc + 255) / 256), 256, 0, stream>>>(gw, wb16, (size_t)Fc * Fc);
    dim3 gg(NNODES / 16, Fc / 256);
    k_gemm_bf16<<<gg, 128, 0, stream>>>(xb16, wb16, h16, NNODES, Fc, Fc);
    k_deg_init<<<(NNODES + 255) / 256, 256, 0, stream>>>(deg, NNODES);
    k_deg_accum<<<(Ec + 255) / 256, 256, 0, stream>>>(ei, deg, Ec);
    k_agg_selfloop<<<(unsigned)((nx + 255) / 256), 256, 0, stream>>>(h16, deg, agg, nx, Fc);
    k_edge_scatter<<<Ec, 256, 0, stream>>>(ei, h16, deg, agg, Ec, Fc);
    k_pool<<<dim3(Fc / 256, Bc), 256, 0, stream>>>(agg, gb, pooled, Fc);
    k_fc_leaky<<<(Bc * ODc + 255) / 256, 256, 0, stream>>>(pooled, fw, fb, xout, Fc, ODc);
  };
  run_branch(pro1_x, p1_ei, gcn1_w, gcn1_b, fc1_w, fc1_b, x1);
  run_branch(pro2_x, p2_ei, gcn2_w, gcn2_b, fc2_w, fc2_b, x2);

  // ================= ragged pack =================
  const size_t nseq = (size_t)Bc * MAXLENc * TDc;
  k_zero<<<(unsigned)((nseq + 255) / 256), 256, 0, stream>>>(xtA, nseq);
  k_total_len<<<1, Bc, 0, stream>>>(l1s, l1f, l2s, l2f, tot);
  k_seq_pack<<<4 * Bc, LSUBc, 0, stream>>>(m1s, m1f, m2s, m2f, l1s, l1f, l2s, l2f,
                                           red_w, red_b, xtA);

  // ================= transformer encoder (post-norm), NL layers =================
  for (int li = 0; li < NLc; ++li) {
    k_qkv<<<(Bc * MAXLENc * 3 * TDc + 255) / 256, 256, 0, stream>>>(
        xtA, ain_w + (size_t)li * 3 * TDc * TDc, ain_b + (size_t)li * 3 * TDc, qkv);
    k_attn<<<(Bc * NHc * MAXLENc) / 4, 128, 0, stream>>>(qkv, tot, attno);
    k_attnout_ln<<<(Bc * MAXLENc + 255) / 256, 256, 0, stream>>>(
        xtA, attno, aout_w + (size_t)li * TDc * TDc, aout_b + (size_t)li * TDc,
        ln1_w + (size_t)li * TDc, ln1_b + (size_t)li * TDc, xtB);
    k_ffn_ln<<<Bc * MAXLENc, DFFc, 0, stream>>>(
        xtB, ff1_w + (size_t)li * DFFc * TDc, ff1_b + (size_t)li * DFFc,
        ff2_w + (size_t)li * TDc * DFFc, ff2_b + (size_t)li * TDc,
        ln2_w + (size_t)li * TDc, ln2_b + (size_t)li * TDc, xtA);
  }

  // ================= masked mean + final linear =================
  k_masked_mean<<<(Bc * TDc + 255) / 256, 256, 0, stream>>>(xtA, tot, masout);
  k_final<<<1, Bc, 0, stream>>>(x1, x2, masout, fin_w, fin_b, out);
}
